// MultiHeadAttention_13666585936365
// MI455X (gfx1250) — matrix operand order repaired and checked by reading, not compiled, not run on hardware
//
#include <hip/hip_runtime.h>

#define S_LEN  2048
#define DMODEL 1024
#define NHEAD  16
#define HDIM   64
#define BATCH  2
#define MROWS  (BATCH * S_LEN)  // 4096

typedef __attribute__((ext_vector_type(16))) _Float16 v16h;
typedef __attribute__((ext_vector_type(8)))  _Float16 v8h;
typedef __attribute__((ext_vector_type(8)))  float    v8f;

union V16U { v16h v; v8h h[2]; };

static __device__ __forceinline__ v8f wmma_f16(v16h a, v16h b, v8f c) {
  return __builtin_amdgcn_wmma_f32_16x16x32_f16(
      false, a, false, b, (short)0, c, false, false);
}

// LDS byte offset of a generic pointer to __shared__: flat->LDS mapping is
// the low 32 bits of the address (ISA 10.2), so truncation is the hardware rule.
static __device__ __forceinline__ unsigned ldsa(const void* p) {
  return (unsigned)(size_t)p;
}
// gfx1250 async DMA: global -> LDS, 16 bytes per lane, tracked by ASYNCcnt.
static __device__ __forceinline__ void async_b128(unsigned lds_off, const void* g) {
  asm volatile("global_load_async_to_lds_b128 %0, %1, off"
               :: "v"(lds_off), "v"(g) : "memory");
}
static __device__ __forceinline__ void wait_async0() {
  asm volatile("s_wait_asynccnt 0x0" ::: "memory");
}

// XOR-swizzle reductions within each 16-lane half (ds_swizzle group-of-32:
// offset = {xor[4:0], or[4:0]=0, and[4:0]=0x1f}); masks 1,2,4,8 stay in-half.
template <int XM>
static __device__ __forceinline__ float swz_max(float x) {
  float o = __int_as_float(
      __builtin_amdgcn_ds_swizzle(__float_as_int(x), (XM << 10) | 0x1f));
  return fmaxf(x, o);
}
template <int XM>
static __device__ __forceinline__ float swz_add(float x) {
  float o = __int_as_float(
      __builtin_amdgcn_ds_swizzle(__float_as_int(x), (XM << 10) | 0x1f));
  return x + o;
}

// A fragment (16x32 f16): lane = row M; elems 0..7 = K=k0+half*8..+7,
// elems 8..15 = K=k0+16+half*8..+7 (half = lane/16).
static __device__ __forceinline__ v16h load_a_frag(const _Float16* base, int ld,
                                                   int row, int k0, int lane) {
  const int half = lane >> 4;
  const _Float16* p = base + (size_t)row * ld + k0 + half * 8;
  V16U u;
  u.h[0] = *(const v8h*)(p);
  u.h[1] = *(const v8h*)(p + 16);
  return u.v;
}
// B fragment (32x16 f16) from [N][K] row-major (B^T): lane = col N,
// elems 0..7 = K=k0+half*8..+7, elems 8..15 = K=k0+16+half*8..+7.
static __device__ __forceinline__ v16h load_b_frag(const _Float16* base, int ld,
                                                   int nrow, int k0, int lane) {
  const int half = lane >> 4;
  const _Float16* p = base + (size_t)nrow * ld + k0 + half * 8;
  V16U u;
  u.h[0] = *(const v8h*)(p);
  u.h[1] = *(const v8h*)(p + 16);
  return u.v;
}

// ---------------------------------------------------------------------------
// fp32 -> f16 conversion (weights, done once instead of per M-block)
// ---------------------------------------------------------------------------
__global__ __launch_bounds__(256) void cvt_kernel(const float* __restrict__ src,
                                                  _Float16* __restrict__ dst,
                                                  int n) {
  const int i = (blockIdx.x * 256 + threadIdx.x) * 8;
  if (i + 8 <= n) {
    float4 a = *(const float4*)(src + i);
    float4 b = *(const float4*)(src + i + 4);
    v8h h;
    h[0] = (_Float16)a.x; h[1] = (_Float16)a.y;
    h[2] = (_Float16)a.z; h[3] = (_Float16)a.w;
    h[4] = (_Float16)b.x; h[5] = (_Float16)b.y;
    h[6] = (_Float16)b.z; h[7] = (_Float16)b.w;
    *(v8h*)(dst + i) = h;
  }
}

// ---------------------------------------------------------------------------
// Projection GEMM: C[4096][1024] = A[4096][1024] * W16[1024][1024]^T + bias.
// Double-buffered LDS tiles; B (and A in MODE 2) staged with async-to-LDS.
// MODE 0: A fp32 -> f16 head-split [B*H][S][HD]
// MODE 1: A fp32 -> f16 transposed  [B*H][HD][S]
// MODE 2: A f16 (y)  -> fp32 row-major d_out
// ---------------------------------------------------------------------------
template <int MODE>
__global__ __launch_bounds__(256) void gemm_proj_kernel(
    const float* __restrict__ A32, const _Float16* __restrict__ A16,
    const _Float16* __restrict__ W16, const float* __restrict__ bias,
    _Float16* __restrict__ dst16, float* __restrict__ dst32) {
  __shared__ __align__(16) _Float16 As[2][64 * 32];   // 2 x 4 KB
  __shared__ __align__(16) _Float16 Bs[2][128 * 32];  // 2 x 8 KB

  const int tid  = threadIdx.x;
  const int lane = tid & 31;
  const int l15  = lane & 15;
  const int rb   = (lane >> 4) * 8;
  const int wave = tid >> 5;
  const int wm   = wave & 3;
  const int wn   = wave >> 2;
  const int M0   = blockIdx.y * 64;
  const int N0   = blockIdx.x * 128;

  // staging lambdas ---------------------------------------------------------
  auto stage = [&](int buf, int k0) {
    {  // B tile 128x32 f16 via async DMA (2 x b128 per thread)
      const int r = tid >> 1;
      const int c = (tid & 1) << 4;
      const _Float16* src = W16 + (size_t)(N0 + r) * DMODEL + k0 + c;
      _Float16* d = &Bs[buf][r * 32 + c];
      async_b128(ldsa(d), src);
      async_b128(ldsa(d + 8), src + 8);
    }
    const int r = tid >> 2;
    const int c = (tid & 3) << 3;
    if (MODE == 2) {  // A already f16: async DMA
      async_b128(ldsa(&As[buf][r * 32 + c]),
                 A16 + (size_t)(M0 + r) * DMODEL + k0 + c);
    } else {          // A fp32: convert in registers, ds_store
      const float* src = A32 + (size_t)(M0 + r) * DMODEL + k0 + c;
      float4 f0 = *(const float4*)(src);
      float4 f1 = *(const float4*)(src + 4);
      v8h hv;
      hv[0] = (_Float16)f0.x; hv[1] = (_Float16)f0.y;
      hv[2] = (_Float16)f0.z; hv[3] = (_Float16)f0.w;
      hv[4] = (_Float16)f1.x; hv[5] = (_Float16)f1.y;
      hv[6] = (_Float16)f1.z; hv[7] = (_Float16)f1.w;
      *(v8h*)&As[buf][r * 32 + c] = hv;
    }
  };

  v8f acc[4];
#pragma unroll
  for (int j = 0; j < 4; ++j)
#pragma unroll
    for (int e = 0; e < 8; ++e) acc[j][e] = 0.0f;

  stage(0, 0);  // prologue
  for (int k0 = 0; k0 < DMODEL; k0 += 32) {
    const int buf = (k0 >> 5) & 1;
    wait_async0();
    __syncthreads();  // tile `buf` complete for all waves; prev compute done
    if (k0 + 32 < DMODEL) stage(1 - buf, k0 + 32);  // overlap next tile

    v16h aF = load_a_frag(As[buf], 32, wm * 16 + l15, 0, lane);
#pragma unroll
    for (int j = 0; j < 4; ++j) {
      v16h bF = load_b_frag(Bs[buf], 32, wn * 64 + j * 16 + l15, 0, lane);
      acc[j] = wmma_f16(aF, bF, acc[j]);
    }
  }

  // epilogue: bias + scatter per MODE (C layout: lane = col, rows rb..rb+7)
#pragma unroll
  for (int j = 0; j < 4; ++j) {
    const int n   = N0 + wn * 64 + j * 16 + l15;
    const float bv = bias[n];
#pragma unroll
    for (int r = 0; r < 8; ++r) {
      const int m     = M0 + wm * 16 + rb + r;
      const float val = acc[j][r] + bv;
      const int bidx = m >> 11;
      const int sidx = m & (S_LEN - 1);
      const int hidx = n >> 6;
      const int didx = n & (HDIM - 1);
      if (MODE == 0) {
        dst16[((size_t)(bidx * NHEAD + hidx) * S_LEN + sidx) * HDIM + didx] =
            (_Float16)val;
      } else if (MODE == 1) {
        dst16[((size_t)(bidx * NHEAD + hidx) * HDIM + didx) * S_LEN + sidx] =
            (_Float16)val;
      } else {
        dst32[(size_t)m * DMODEL + n] = val;
      }
    }
  }
}

// ---------------------------------------------------------------------------
// Flash attention: block = 64 q-rows of one (b,h); key blocks of 128 with
// double-buffered async K tiles; online softmax with ds_swizzle reductions;
// mask staged as a bitmask (1 bit/elem) via coalesced loads.
// ---------------------------------------------------------------------------
__global__ __launch_bounds__(256) void attn_kernel(
    const _Float16* __restrict__ qh, const _Float16* __restrict__ kh,
    const _Float16* __restrict__ vT, const int* __restrict__ mask,
    _Float16* __restrict__ yb) {
  __shared__ __align__(16) _Float16 Qs[64 * 64];       // 8 KB
  __shared__ __align__(16) _Float16 Ks[2][128 * 64];   // 2 x 16 KB
  __shared__ __align__(16) _Float16 Ps[64 * 128];      // 16 KB
  __shared__ __align__(16) unsigned Mbits[64 * 4];     // 1 KB bitmask tile
  __shared__ float redS[2 * 64];                       // cross-wave combine
  __shared__ float m_run[64];
  __shared__ float l_run[64];
  __shared__ float alpha_s[64];

  const int tid  = threadIdx.x;
  const int lane = tid & 31;
  const int l15  = lane & 15;
  const int rb   = (lane >> 4) * 8;
  const int wave = tid >> 5;
  const int wm   = wave & 3;
  const int wn   = wave >> 2;

  const int q0 = blockIdx.x * 64;
  const int bh = blockIdx.y;
  const int b  = bh >> 4;
  const int h  = bh & (NHEAD - 1);

  const _Float16* qhh = qh + (size_t)bh * S_LEN * HDIM;
  const _Float16* khh = kh + (size_t)bh * S_LEN * HDIM;
  const _Float16* vTh = vT + (size_t)bh * HDIM * S_LEN;
  const int*      mb  = mask + (size_t)b * S_LEN * S_LEN;

  auto stage_k = [&](int buf, int kb) {  // 128x64 f16, 4 x b128 per thread
    const _Float16* src = khh + (size_t)kb * HDIM + tid * 32;
    _Float16* d = &Ks[buf][tid * 32];
#pragma unroll
    for (int i = 0; i < 4; ++i) async_b128(ldsa(d + i * 8), src + i * 8);
  };

  // prologue: async Q tile + first K tile
  {
    const _Float16* src = qhh + (size_t)q0 * HDIM + tid * 16;
    async_b128(ldsa(&Qs[tid * 16]), src);
    async_b128(ldsa(&Qs[tid * 16 + 8]), src + 8);
  }
  stage_k(0, 0);
  if (tid < 64) { m_run[tid] = -3.0e38f; l_run[tid] = 0.0f; }

  v8f acc2[2];
#pragma unroll
  for (int jj = 0; jj < 2; ++jj)
#pragma unroll
    for (int e = 0; e < 8; ++e) acc2[jj][e] = 0.0f;

  wait_async0();  // own Q + K0 complete (B1 below makes it block-wide)

  for (int kb = 0; kb < S_LEN; kb += 128) {
    const int buf = (kb >> 7) & 1;

    // stage mask bitmask: each thread packs 32 cols of one row (coalesced)
    {
      const int r = tid >> 2;
      const int c = (tid & 3) << 5;
      const int* src = mb + (size_t)(q0 + r) * S_LEN + kb + c;
      unsigned w = 0u;
#pragma unroll
      for (int i = 0; i < 32; i += 4) {
        int4 m4 = *(const int4*)(src + i);
        w |= (m4.x == 0 ? 1u : 0u) << (i + 0);
        w |= (m4.y == 0 ? 1u : 0u) << (i + 1);
        w |= (m4.z == 0 ? 1u : 0u) << (i + 2);
        w |= (m4.w == 0 ? 1u : 0u) << (i + 3);
      }
      Mbits[tid] = w;
    }
    if (kb + 128 < S_LEN) stage_k(1 - buf, kb + 128);  // overlap next K tile
    __syncthreads();  // [B1] Mbits + current K tile visible block-wide

    // GEMM1: scores 16q x 64k per wave (K = HDIM = 64)
    v8f sc[4];
#pragma unroll
    for (int j = 0; j < 4; ++j)
#pragma unroll
      for (int e = 0; e < 8; ++e) sc[j][e] = 0.0f;
#pragma unroll
    for (int kk = 0; kk < HDIM; kk += 32) {
      v16h aF = load_a_frag(Qs, 64, wm * 16 + l15, kk, lane);
#pragma unroll
      for (int j = 0; j < 4; ++j) {
        v16h bF = load_b_frag(Ks[buf], 64, wn * 64 + j * 16 + l15, kk, lane);
        sc[j] = wmma_f16(aF, bF, sc[j]);
      }
    }

    // scale + mask (bitmask) + per-row max via swizzle reduction
#pragma unroll
    for (int r = 0; r < 8; ++r) {
      const int rt = wm * 16 + rb + r;
      const unsigned long long wb =
          *(const unsigned long long*)&Mbits[rt * 4 + wn * 2];
      float pm = -3.0e38f;
#pragma unroll
      for (int j = 0; j < 4; ++j) {
        float v = sc[j][r] * 0.125f;
        if ((wb >> (j * 16 + l15)) & 1ull) v = -1.0e9f;
        sc[j][r] = v;
        pm = fmaxf(pm, v);
      }
      pm = swz_max<1>(pm); pm = swz_max<2>(pm);
      pm = swz_max<4>(pm); pm = swz_max<8>(pm);
      if (l15 == 0) redS[wn * 64 + rt] = pm;  // lane0: rows rb=0, lane16: rb=8
    }
    __syncthreads();  // [B2]

    if (tid < 64) {  // combine 2 wave-columns, online-softmax stats
      const float mx = fmaxf(redS[tid], redS[64 + tid]);
      const float mo = m_run[tid];
      const float mn = fmaxf(mo, mx);
      const float al = __expf(mo - mn);
      alpha_s[tid] = al;
      m_run[tid]   = mn;
      l_run[tid]  *= al;
    }
    __syncthreads();  // [B3]

    // P = exp(sc - m_new) -> Ps f16; row sums via swizzle; rescale y accs
#pragma unroll
    for (int r = 0; r < 8; ++r) {
      const int rt = wm * 16 + rb + r;
      const float mrow = m_run[rt];
      float ps = 0.0f;
#pragma unroll
      for (int j = 0; j < 4; ++j) {
        const float p = __expf(sc[j][r] - mrow);
        ps += p;
        Ps[rt * 128 + wn * 64 + j * 16 + l15] = (_Float16)p;
      }
      ps = swz_add<1>(ps); ps = swz_add<2>(ps);
      ps = swz_add<4>(ps); ps = swz_add<8>(ps);
      if (l15 == 0) redS[wn * 64 + rt] = ps;
    }
#pragma unroll
    for (int jj = 0; jj < 2; ++jj)
#pragma unroll
      for (int r = 0; r < 8; ++r)
        acc2[jj][r] *= alpha_s[wm * 16 + rb + r];
    __syncthreads();  // [B4]

    if (tid < 64) l_run[tid] += redS[tid] + redS[64 + tid];

    // GEMM2: y += P(64x128) * V; B-frags straight from vT (K contiguous)
#pragma unroll
    for (int kk = 0; kk < 128; kk += 32) {
      v16h aF = load_a_frag(Ps, 128, wm * 16 + l15, kk, lane);
#pragma unroll
      for (int jj = 0; jj < 2; ++jj) {
        v16h bF = load_b_frag(vTh, S_LEN, wn * 32 + jj * 16 + l15, kb + kk, lane);
        acc2[jj] = wmma_f16(aF, bF, acc2[jj]);
      }
    }
    wait_async0();  // own next-tile asyncs done; [B1] of next iter syncs all
  }

  // finalize: divide by row sum, write y f16 in [B][S][D]
#pragma unroll
  for (int jj = 0; jj < 2; ++jj) {
#pragma unroll
    for (int r = 0; r < 8; ++r) {
      const int rt = wm * 16 + rb + r;
      const float denom = fmaxf(l_run[rt], 1e-20f);
      const float val = acc2[jj][r] / denom;
      const int sidx = q0 + rt;
      const int dcol = wn * 32 + jj * 16 + l15;
      yb[((size_t)(b * S_LEN + sidx)) * DMODEL + h * HDIM + dcol] =
          (_Float16)val;
    }
  }
}

extern "C" void kernel_launch(void* const* d_in, const int* in_sizes, int n_in,
                              void* d_out, int out_size, void* d_ws,
                              size_t ws_size, hipStream_t stream) {
  (void)in_sizes; (void)n_in; (void)out_size; (void)ws_size;
  const float* q    = (const float*)d_in[0];
  const float* k    = (const float*)d_in[1];
  const float* v    = (const float*)d_in[2];
  const int*   mask = (const int*)d_in[3];
  const float* Wq   = (const float*)d_in[4];
  const float* bq   = (const float*)d_in[5];
  const float* Wk   = (const float*)d_in[6];
  const float* bk   = (const float*)d_in[7];
  const float* Wv   = (const float*)d_in[8];
  const float* bv   = (const float*)d_in[9];
  const float* Wo   = (const float*)d_in[10];
  const float* bo   = (const float*)d_in[11];
  float* out = (float*)d_out;

  const size_t W_ELEMS = (size_t)DMODEL * DMODEL;  // 1M
  const size_t X_ELEMS = (size_t)MROWS * DMODEL;   // 4M
  _Float16* wq16 = (_Float16*)d_ws;
  _Float16* wk16 = wq16 + W_ELEMS;
  _Float16* wv16 = wk16 + W_ELEMS;
  _Float16* wo16 = wv16 + W_ELEMS;
  _Float16* qh   = wo16 + W_ELEMS;
  _Float16* kh   = qh + X_ELEMS;
  _Float16* vT   = kh + X_ELEMS;
  _Float16* yb   = vT + X_ELEMS;  // total 20M f16 = 40 MB workspace

  const int cvtGrid = (int)(W_ELEMS / (256 * 8));  // 512
  cvt_kernel<<<cvtGrid, 256, 0, stream>>>(Wq, wq16, (int)W_ELEMS);
  cvt_kernel<<<cvtGrid, 256, 0, stream>>>(Wk, wk16, (int)W_ELEMS);
  cvt_kernel<<<cvtGrid, 256, 0, stream>>>(Wv, wv16, (int)W_ELEMS);
  cvt_kernel<<<cvtGrid, 256, 0, stream>>>(Wo, wo16, (int)W_ELEMS);

  dim3 gGrid(DMODEL / 128, MROWS / 64);  // (8, 64)
  gemm_proj_kernel<0><<<gGrid, 256, 0, stream>>>(q, nullptr, wq16, bq, qh, nullptr);
  gemm_proj_kernel<0><<<gGrid, 256, 0, stream>>>(k, nullptr, wk16, bk, kh, nullptr);
  gemm_proj_kernel<1><<<gGrid, 256, 0, stream>>>(v, nullptr, wv16, bv, vT, nullptr);

  attn_kernel<<<dim3(S_LEN / 64, BATCH * NHEAD), 256, 0, stream>>>(
      qh, kh, vT, mask, yb);

  gemm_proj_kernel<2><<<gGrid, 256, 0, stream>>>(nullptr, yb, wo16, bo, nullptr, out);
}
